// VQVAEMapper_1245540516311
// MI455X (gfx1250) — matrix operand order repaired and checked by reading, not compiled, not run on hardware
//
#include <hip/hip_runtime.h>
#include <hip/hip_bf16.h>

// VQ-VAE codebook mapper for MI455X (gfx1250, wave32).
//   out[n] = argmin_k ||x_n - c_k||^2  =  argmin_k ( ||c_k||^2 - 2 x_n . c_k )
// Compute-bound GEMM (34.4 GFLOP vs ~10.5 MB traffic) -> WMMA f16x3 fp32-emulated
// dot products (hi/lo split, ~22 mantissa bits) with fused per-lane argmin.
// Codebook chunks are prefetched with CDNA5 GLOBAL_LOAD_ASYNC_TO_LDS_B128
// (ASYNCcnt) so the global fetch of chunk n+1 overlaps the WMMAs of chunk n.

typedef __attribute__((ext_vector_type(16))) _Float16 v16h;
typedef __attribute__((ext_vector_type(8)))  _Float16 v8h;
typedef __attribute__((ext_vector_type(8)))  float    v8f;
typedef __attribute__((ext_vector_type(4)))  float    v4f;

#define DDIM 64            // feature dimension (fixed by problem)
#define CHUNK 64           // codebook rows staged in LDS per iteration
#define TILES (CHUNK / 16) // 4 A-tiles of 16 codes
#define THREADS 256        // 8 waves
#define WAVES (THREADS / 32)
#define COLS_PER_WAVE 32   // 2 column tiles of 16 latents per wave
#define COLS_PER_WG (WAVES * COLS_PER_WAVE) // 256 latents per workgroup

union V16U { v16h v; v8h h[2]; };

// Issue this thread's share (64 B) of an async global->LDS chunk copy.
// IOFFSET is added to BOTH the LDS address and the global address (ISA 15.18.3),
// so one base pair serves all four b128 transfers.
__device__ __forceinline__ void async_stage(const float* gsrc, float* lds_dst) {
    unsigned long long ga = (unsigned long long)gsrc;
    unsigned ldsa = (unsigned)(unsigned long long)lds_dst; // low 32b = LDS offset
    asm volatile("global_load_async_to_lds_b128 %0, %1, off"           :: "v"(ldsa), "v"(ga) : "memory");
    asm volatile("global_load_async_to_lds_b128 %0, %1, off offset:16" :: "v"(ldsa), "v"(ga) : "memory");
    asm volatile("global_load_async_to_lds_b128 %0, %1, off offset:32" :: "v"(ldsa), "v"(ga) : "memory");
    asm volatile("global_load_async_to_lds_b128 %0, %1, off offset:48" :: "v"(ldsa), "v"(ga) : "memory");
}

__global__ __launch_bounds__(THREADS)
void vq_argmin_wmma(const float* __restrict__ latents,
                    const float* __restrict__ codebook,
                    int* __restrict__ out,
                    int K)
{
    __shared__ float    sRaw[CHUNK * DDIM];   // 16 KB raw fp32 chunk (async target)
    __shared__ _Float16 sAhi[CHUNK * DDIM];   // 8 KB
    __shared__ _Float16 sAlo[CHUNK * DDIM];   // 8 KB
    __shared__ float    sC2[CHUNK];           // 256 B exact fp32 ||c||^2
    __shared__ float    sPart[THREADS];       // 1 KB

    const int tid     = threadIdx.x;
    const int lane    = tid & 31;
    const int wave    = tid >> 5;
    const int halfSel = lane >> 4;   // 0: lanes 0-15, 1: lanes 16-31
    const int l16     = lane & 15;

    const int colBase = blockIdx.x * COLS_PER_WG + wave * COLS_PER_WAVE;

    // ---- Load this wave's 32 latent columns as B operands (f16 hi + lo), kept in VGPRs.
    // B operand for V_WMMA_F32_16X16X32_F16 (32x16, KxN), contiguous K order used here: lane = column N (mod 16);
    // lanes 0-15 hold K=0..15, lanes 16-31 hold K=16..31; half j of v16h <-> K = kb2+j.
    V16U bhi[2][2], blo[2][2];       // [column tile][K segment: K 0..31 / 32..63]
    const int kb2 = halfSel * 16;
    #pragma unroll
    for (int ct = 0; ct < 2; ++ct) {
        const int col = colBase + ct * 16 + l16;
        const float* base = latents + (size_t)col * DDIM;
        #pragma unroll
        for (int seg = 0; seg < 2; ++seg) {
            const v4f* vp = reinterpret_cast<const v4f*>(base + seg * 32 + kb2);
            float xv[16];
            #pragma unroll
            for (int q = 0; q < 4; ++q) {
                v4f f = vp[q];
                #pragma unroll
                for (int e = 0; e < 4; ++e) xv[4 * q + e] = f[e];
            }
            #pragma unroll
            for (int j = 0; j < 16; ++j) {
                float x = xv[j];
                _Float16 h = (_Float16)x;          // hi: 11 mantissa bits
                float r = x - (float)h;            // exact residual
                bhi[ct][seg].v[j] = h;
                blo[ct][seg].v[j] = (_Float16)r;   // lo: ~11 more bits
            }
        }
    }

    float bestV[2] = { 3.4e38f, 3.4e38f };
    int   bestI[2] = { 0, 0 };

    // A operand for 16x16x32 f16 as loaded here (16x32, MxK): lane = row M (mod 16); lanes 0-15
    // hold K 0..15, lanes 16-31 hold K 16..31 (same order as B); a_kb = halfSel*8 is the D row base.
    const int a_kb = halfSel * 8;

    // Thread's share of a chunk: 16 consecutive floats at linear offset tid*16.
    const int myOff = tid * 16;                  // CHUNK*DDIM/THREADS == 16

    // ---- Prologue: async-stage chunk 0
    async_stage(codebook + myOff, sRaw + myOff);

    for (int cb = 0; cb < K; cb += CHUNK) {
        asm volatile("s_wait_asynccnt 0" ::: "memory");
        __syncthreads();  // raw chunk visible to all; prior compute done (hi/lo reusable)

        // ---- Convert raw fp32 -> f16 hi/lo + partial ||c||^2 (4 threads per code row)
        {
            const int r  = tid >> 2;
            const int c0 = (tid & 3) << 4;
            const v4f* vp = reinterpret_cast<const v4f*>(sRaw + r * DDIM + c0);
            float ss = 0.f;
            #pragma unroll
            for (int q = 0; q < 4; ++q) {
                v4f f = vp[q];
                #pragma unroll
                for (int e = 0; e < 4; ++e) {
                    float x = f[e];
                    _Float16 h = (_Float16)x;
                    float rr = x - (float)h;
                    sAhi[r * DDIM + c0 + 4 * q + e] = h;
                    sAlo[r * DDIM + c0 + 4 * q + e] = (_Float16)rr;
                    ss += x * x;
                }
            }
            sPart[tid] = ss;
        }
        __syncthreads();  // raw fully consumed; sPart complete

        if (tid < CHUNK)
            sC2[tid] = (sPart[4 * tid] + sPart[4 * tid + 1])
                     + (sPart[4 * tid + 2] + sPart[4 * tid + 3]);

        // ---- Prefetch next chunk (overlaps the WMMA compute below)
        if (cb + CHUNK < K)
            async_stage(codebook + (size_t)(cb + CHUNK) * DDIM + myOff, sRaw + myOff);
        __syncthreads();  // sC2 ready; safe to compute while async fills sRaw

        // ---- Compute: 4 code-tiles x 2 column-tiles x 6 WMMAs (hh,hh,hl,hl,lh,lh)
        #pragma unroll
        for (int t = 0; t < TILES; ++t) {
            const int r = t * 16 + l16;
            const v8h* ph = reinterpret_cast<const v8h*>(sAhi + r * DDIM);
            const v8h* pl = reinterpret_cast<const v8h*>(sAlo + r * DDIM);
            V16U ah0, ah1, al0, al1;                 // [K 0..31], [K 32..63]
            ah0.h[0] = ph[2 * halfSel];     ah0.h[1] = ph[2 * halfSel + 1];
            ah1.h[0] = ph[2 * halfSel + 4]; ah1.h[1] = ph[2 * halfSel + 5];
            al0.h[0] = pl[2 * halfSel];     al0.h[1] = pl[2 * halfSel + 1];
            al1.h[0] = pl[2 * halfSel + 4]; al1.h[1] = pl[2 * halfSel + 5];

            // ||c||^2 for this lane's 8 codes (M = a_kb + v)
            const v4f* pc2 = reinterpret_cast<const v4f*>(sC2 + t * 16 + a_kb);
            v4f c2a = pc2[0], c2b = pc2[1];
            const int idx0 = cb + t * 16 + a_kb;

            #pragma unroll
            for (int ct = 0; ct < 2; ++ct) {
                v8f acc = {};
                acc = __builtin_amdgcn_wmma_f32_16x16x32_f16(false, ah0.v, false, bhi[ct][0].v, (short)0, acc, false, false);
                acc = __builtin_amdgcn_wmma_f32_16x16x32_f16(false, ah1.v, false, bhi[ct][1].v, (short)0, acc, false, false);
                acc = __builtin_amdgcn_wmma_f32_16x16x32_f16(false, ah0.v, false, blo[ct][0].v, (short)0, acc, false, false);
                acc = __builtin_amdgcn_wmma_f32_16x16x32_f16(false, ah1.v, false, blo[ct][1].v, (short)0, acc, false, false);
                acc = __builtin_amdgcn_wmma_f32_16x16x32_f16(false, al0.v, false, bhi[ct][0].v, (short)0, acc, false, false);
                acc = __builtin_amdgcn_wmma_f32_16x16x32_f16(false, al1.v, false, bhi[ct][1].v, (short)0, acc, false, false);

                // C layout: acc[v] = (code M = a_kb+v, latent N = l16); running argmin,
                // ascending code order + strict '<' => first-min tie rule like jnp.argmin.
                #pragma unroll
                for (int v = 0; v < 8; ++v) {
                    float c2v = (v < 4) ? c2a[v] : c2b[v - 4];
                    float s = c2v - 2.0f * acc[v];   // ||x||^2 dropped (argmin-invariant)
                    if (s < bestV[ct]) { bestV[ct] = s; bestI[ct] = idx0 + v; }
                }
            }
        }
        // no trailing barrier: next iteration's wait+barrier orders compute(n)
        // before convert(n+1) rewrites sAhi/sAlo.
    }

    // ---- Merge lane halves (same latent column N = l16) and write int32 indices
    #pragma unroll
    for (int ct = 0; ct < 2; ++ct) {
        float ov = __shfl_xor(bestV[ct], 16, 32);
        int   oi = __shfl_xor(bestI[ct], 16, 32);
        if (ov < bestV[ct] || (ov == bestV[ct] && oi < bestI[ct])) {
            bestV[ct] = ov; bestI[ct] = oi;
        }
        if (halfSel == 0) out[colBase + ct * 16 + l16] = bestI[ct];
    }
}

extern "C" void kernel_launch(void* const* d_in, const int* in_sizes, int n_in,
                              void* d_out, int out_size, void* d_ws, size_t ws_size,
                              hipStream_t stream) {
    const float* latents  = (const float*)d_in[0];   // (32,1024,64) fp32
    const float* codebook = (const float*)d_in[1];   // (8192,64)    fp32
    int* out = (int*)d_out;                          // (32,1024) int32 indices

    const int Nq = in_sizes[0] / DDIM;               // 32768
    const int K  = in_sizes[1] / DDIM;               // 8192
    const int grid = Nq / COLS_PER_WG;               // 128 workgroups

    vq_argmin_wmma<<<grid, THREADS, 0, stream>>>(latents, codebook, out, K);
}
